// ReputationPredictor_4252017623388
// MI455X (gfx1250) — matrix operand order repaired and checked by reading, not compiled, not run on hardware
//
#include <hip/hip_runtime.h>
#include <hip/hip_bf16.h>
#include <math.h>

typedef __attribute__((ext_vector_type(16))) _Float16 v16h;
typedef __attribute__((ext_vector_type(8)))  float    v8f;
typedef _Float16 h16;

// Problem constants (from the reference)
#define DN   1024   // neighbors (batch)
#define KS   128    // seq len
#define ED   384    // edge attr dim
#define HID  128    // hidden
#define NL   2      // layers
#define NH   4      // heads
#define HSZ  32     // head dim
#define FFD  128    // ff dim

// LDS leading dims (padded to stagger banks, keep 16B alignment for f16 rows)
#define LDH  136    // f16 buffers: 136*2 = 272B rows (16B aligned, 4-bank stagger)
#define LDF  132    // f32 score buffer
#define SMEM_BYTES (6 * KS * LDH * 2 + KS * LDF * 4)   // 276480 B < 320KB

// ---------------------------------------------------------------------------
// WMMA helpers — explicit CDNA5 16x16x32 f16 fragment layouts (ISA 7.12.2)
// ---------------------------------------------------------------------------
__device__ __forceinline__ v8f vzero8() {
  v8f z;
#pragma unroll
  for (int i = 0; i < 8; ++i) z[i] = 0.0f;
  return z;
}

__device__ __forceinline__ v8f wmma16(v16h a, v16h b, v8f c) {
  // (neg_a, A, neg_b, B, c_mod, C, reuse_a, reuse_b)
  return __builtin_amdgcn_wmma_f32_16x16x32_f16(false, a, false, b, (short)0, c,
                                                false, false);
}

// A (16x32, MxK) from f16 row-major [M][ld]: lane m = lane%16;
// elems 0..7 -> K = (hi?8:0)+i ; elems 8..15 -> K = (hi?24:16)+i
__device__ __forceinline__ v16h load_a16(const h16* base, int ld, int m0, int k0,
                                         int lane) {
  const h16* p = base + (size_t)(m0 + (lane & 15)) * ld + k0;
  const int o0 = (lane & 16) ? 8 : 0;
  const int o1 = (lane & 16) ? 24 : 16;
  v16h r;
#pragma unroll
  for (int i = 0; i < 8; ++i) r[i] = p[o0 + i];
#pragma unroll
  for (int i = 0; i < 8; ++i) r[8 + i] = p[o1 + i];
  return r;
}

// Same A layout but sourced from f32 (global edge_attrs), converted on load.
__device__ __forceinline__ v16h load_a32(const float* base, int ld, int m0, int k0,
                                         int lane) {
  const float* p = base + (size_t)(m0 + (lane & 15)) * ld + k0;
  const int o0 = (lane & 16) ? 8 : 0;
  const int o1 = (lane & 16) ? 24 : 16;
  v16h r;
#pragma unroll
  for (int i = 0; i < 8; ++i) r[i] = (h16)p[o0 + i];
#pragma unroll
  for (int i = 0; i < 8; ++i) r[8 + i] = (h16)p[o1 + i];
  return r;
}

// B (32x16, KxN) where source is [N][ld] row-major with K contiguous
// (i.e. weight[out][in], or LDS k/vT buffers). Lane n = lane%16;
// elems 0..7 -> K = k0 + (hi?8:0)+i ; elems 8..15 -> K = k0 + (hi?24:16)+i.
__device__ __forceinline__ v16h load_b16(const h16* w, int ld, int n0, int k0,
                                         int lane) {
  const h16* p = w + (size_t)(n0 + (lane & 15)) * ld + k0;
  const int o0 = (lane & 16) ? 8 : 0;
  const int o1 = (lane & 16) ? 24 : 16;
  v16h r;
#pragma unroll
  for (int i = 0; i < 8; ++i) r[i] = p[o0 + i];
#pragma unroll
  for (int i = 0; i < 8; ++i) r[8 + i] = p[o1 + i];
  return r;
}

// ---------------------------------------------------------------------------
// Pass 0: deterministic prefix-scan to find, per neighbor slot, the index of
// its tgt->n edge (slot order = edge order of tgt edges, matching reference).
// ---------------------------------------------------------------------------
__global__ __launch_bounds__(1024) void find_tgt_edges(
    const int* __restrict__ edge_index, int E, const int* __restrict__ tgt_ptr,
    int* __restrict__ tgtEdge) {
  __shared__ int fl[1024];
  __shared__ int running;
  const int tid = threadIdx.x;
  const int tgt = tgt_ptr[0];
  const int* src = edge_index;  // row 0 of (2, E)
  if (tid == 0) running = 0;
  __syncthreads();
  for (int base = 0; base < E; base += 1024) {
    const int e = base + tid;
    const int f = (e < E && src[e] == tgt) ? 1 : 0;
    fl[tid] = f;
    __syncthreads();
#pragma unroll
    for (int off = 1; off < 1024; off <<= 1) {
      int v = fl[tid];
      int add = (tid >= off) ? fl[tid - off] : 0;
      __syncthreads();
      fl[tid] = v + add;
      __syncthreads();
    }
    if (f) {
      int slot = running + fl[tid] - 1;
      if (slot < DN) tgtEdge[slot] = e;
    }
    __syncthreads();
    if (tid == 0) running += fl[1023];
    __syncthreads();
  }
}

// ---------------------------------------------------------------------------
// Weight f32 -> f16 conversion (layout preserved; [out][in] == B operand)
// ---------------------------------------------------------------------------
__global__ void f32_to_f16(const float* __restrict__ in, h16* __restrict__ out,
                           int n) {
  int i = blockIdx.x * blockDim.x + threadIdx.x;
  if (i < n) out[i] = (h16)in[i];
}

// ---------------------------------------------------------------------------
// Row LayerNorm over HID cols of an f16 LDS matrix (threads 0..127 own rows)
// ---------------------------------------------------------------------------
__device__ __forceinline__ void ln_rows(h16* sh, const float* __restrict__ g,
                                        const float* __restrict__ b, int tid) {
  if (tid < KS) {
    h16* r = sh + (size_t)tid * LDH;
    float m = 0.f;
#pragma unroll 4
    for (int j = 0; j < HID; ++j) m += (float)r[j];
    m *= (1.0f / HID);
    float v = 0.f;
#pragma unroll 4
    for (int j = 0; j < HID; ++j) {
      float d = (float)r[j] - m;
      v += d * d;
    }
    v *= (1.0f / HID);
    const float inv = rsqrtf(v + 1e-5f);
#pragma unroll 4
    for (int j = 0; j < HID; ++j)
      r[j] = (h16)(((float)r[j] - m) * inv * g[j] + b[j]);
  }
}

// ---------------------------------------------------------------------------
// Main kernel: one workgroup per neighbor; whole transformer in LDS via WMMA.
// 256 threads = 8 waves; wave w owns M-tile rows [16w, 16w+16).
// ---------------------------------------------------------------------------
__global__ __launch_bounds__(256) void neighbor_transformer(
    const float* __restrict__ edge_attrs, const int* __restrict__ tgtEdge,
    const h16* __restrict__ w_proj, const float* __restrict__ proj_b,
    const float* __restrict__ marker, const h16* __restrict__ w_in,
    const float* __restrict__ b_in, const h16* __restrict__ w_out,
    const float* __restrict__ b_out, const float* __restrict__ ln1_g,
    const float* __restrict__ ln1_b, const h16* __restrict__ w_ff1,
    const float* __restrict__ b_ff1, const h16* __restrict__ w_ff2,
    const float* __restrict__ b_ff2, const float* __restrict__ ln2_g,
    const float* __restrict__ ln2_b, float* __restrict__ pooled) {
  extern __shared__ __align__(16) char smem[];
  h16* sh = (h16*)smem;            // activations  [KS][LDH]
  h16* sq = sh + KS * LDH;         // Q            [KS][LDH]
  h16* sk = sq + KS * LDH;         // K            [KS][LDH]
  h16* sv = sk + KS * LDH;         // V transposed [HID][LDH] (dim-major)
  h16* sat = sv + KS * LDH;        // att probs / FF intermediate
  h16* so = sat + KS * LDH;        // attention output
  float* ss = (float*)(so + KS * LDH);  // scores [KS][LDF] f32

  const int tid = threadIdx.x;
  const int lane = tid & 31;
  const int wv = tid >> 5;
  const int m0 = wv * 16;
  const int ncol = lane & 15;
  const int mrow = (lane >> 4) << 3;  // D-tile: VGPR r -> M = r + (hi?8:0)
  const int s = blockIdx.x;

  // ---- Phase 0: projection  h = rows @ proj_w^T + b ----
  {
    const float* rows = edge_attrs + (size_t)tgtEdge[s] * ED;  // [KS][ED]
    v8f acc[8];
#pragma unroll
    for (int i = 0; i < 8; ++i) acc[i] = vzero8();
    for (int k0 = 0; k0 < ED; k0 += 32) {
      // pull the next K-chunk's cacheline toward the WGP while WMMAs run
      if (k0 + 32 < ED)
        __builtin_prefetch(rows + (size_t)(m0 + (lane & 15)) * ED + k0 + 32, 0, 3);
      v16h a = load_a32(rows, ED, m0, k0, lane);
#pragma unroll
      for (int nt = 0; nt < 8; ++nt) {
        v16h b = load_b16(w_proj, ED, nt * 16, k0, lane);
        acc[nt] = wmma16(a, b, acc[nt]);
      }
    }
#pragma unroll
    for (int nt = 0; nt < 8; ++nt) {
      const int n = nt * 16 + ncol;
      const float bias = proj_b[n];
#pragma unroll
      for (int r = 0; r < 8; ++r) {
        const int m = m0 + mrow + r;
        sh[(size_t)m * LDH + n] = (h16)(acc[nt][r] + bias);
      }
    }
  }
  __syncthreads();
  // first-token marker on row 0 (uniform, no divergence in the GEMM epilogue)
  if (tid < HID) sh[tid] = (h16)((float)sh[tid] + marker[tid]);
  __syncthreads();

  // ---- Transformer layers ----
  for (int l = 0; l < NL; ++l) {
    const h16* Win = w_in + (size_t)l * 3 * HID * HID;
    const float* Bin = b_in + l * 3 * HID;
    const h16* Wout = w_out + (size_t)l * HID * HID;
    const float* Bout = b_out + l * HID;
    const h16* Wf1 = w_ff1 + (size_t)l * FFD * HID;
    const float* Bf1 = b_ff1 + l * FFD;
    const h16* Wf2 = w_ff2 + (size_t)l * HID * FFD;
    const float* Bf2 = b_ff2 + l * HID;

    // QKV: q,k normal layout; v stored transposed (dim-major) for att@V
    for (int p = 0; p < 3; ++p) {
      const h16* W = Win + (size_t)p * HID * HID;
      v8f acc[8];
#pragma unroll
      for (int i = 0; i < 8; ++i) acc[i] = vzero8();
      for (int k0 = 0; k0 < HID; k0 += 32) {
        v16h a = load_a16(sh, LDH, m0, k0, lane);
#pragma unroll
        for (int nt = 0; nt < 8; ++nt) {
          v16h b = load_b16(W, HID, nt * 16, k0, lane);
          acc[nt] = wmma16(a, b, acc[nt]);
        }
      }
      h16* dst = (p == 0) ? sq : sk;
#pragma unroll
      for (int nt = 0; nt < 8; ++nt) {
        const int n = nt * 16 + ncol;
        const float bias = Bin[p * HID + n];
#pragma unroll
        for (int r = 0; r < 8; ++r) {
          const int m = m0 + mrow + r;
          const float val = acc[nt][r] + bias;
          if (p < 2)
            dst[(size_t)m * LDH + n] = (h16)val;
          else
            sv[(size_t)n * LDH + m] = (h16)val;  // transpose store
        }
      }
    }
    __syncthreads();

    // Attention, head by head (hd = 32 -> one WMMA per score tile)
    const float rscale = 0.17677669529663687f;  // 1/sqrt(32)
    for (int hh = 0; hh < NH; ++hh) {
      {  // scores = (q_h @ k_h^T) * rscale -> ss (f32)
        v16h a = load_a16(sq, LDH, m0, hh * HSZ, lane);
#pragma unroll
        for (int nt = 0; nt < 8; ++nt) {
          v16h b = load_b16(sk, LDH, nt * 16, hh * HSZ, lane);
          v8f c = wmma16(a, b, vzero8());
          const int n = nt * 16 + ncol;
#pragma unroll
          for (int r = 0; r < 8; ++r)
            ss[(size_t)(m0 + mrow + r) * LDF + n] = c[r] * rscale;
        }
      }
      __syncthreads();
      if (tid < KS) {  // row softmax -> sat (f16)
        float* sr = ss + (size_t)tid * LDF;
        float mx = -1e30f;
#pragma unroll 4
        for (int j = 0; j < KS; ++j) mx = fmaxf(mx, sr[j]);
        float sum = 0.f;
#pragma unroll 4
        for (int j = 0; j < KS; ++j) {
          float e = __expf(sr[j] - mx);
          sr[j] = e;
          sum += e;
        }
        const float inv = 1.0f / sum;
        h16* ar = sat + (size_t)tid * LDH;
#pragma unroll 4
        for (int j = 0; j < KS; ++j) ar[j] = (h16)(sr[j] * inv);
      }
      __syncthreads();
      {  // o_h = att @ v_h  (B from transposed V: contiguous K)
        v8f acc[2];
        acc[0] = vzero8();
        acc[1] = vzero8();
        for (int k0 = 0; k0 < KS; k0 += 32) {
          v16h a = load_a16(sat, LDH, m0, k0, lane);
#pragma unroll
          for (int nt = 0; nt < 2; ++nt) {
            v16h b = load_b16(sv, LDH, hh * HSZ + nt * 16, k0, lane);
            acc[nt] = wmma16(a, b, acc[nt]);
          }
        }
#pragma unroll
        for (int nt = 0; nt < 2; ++nt) {
          const int n = hh * HSZ + nt * 16 + ncol;
#pragma unroll
          for (int r = 0; r < 8; ++r)
            so[(size_t)(m0 + mrow + r) * LDH + n] = (h16)acc[nt][r];
        }
      }
      __syncthreads();
    }

    // Output projection + residual, then LN1
    {
      v8f acc[8];
#pragma unroll
      for (int i = 0; i < 8; ++i) acc[i] = vzero8();
      for (int k0 = 0; k0 < HID; k0 += 32) {
        v16h a = load_a16(so, LDH, m0, k0, lane);
#pragma unroll
        for (int nt = 0; nt < 8; ++nt) {
          v16h b = load_b16(Wout, HID, nt * 16, k0, lane);
          acc[nt] = wmma16(a, b, acc[nt]);
        }
      }
#pragma unroll
      for (int nt = 0; nt < 8; ++nt) {
        const int n = nt * 16 + ncol;
        const float bias = Bout[n];
#pragma unroll
        for (int r = 0; r < 8; ++r) {
          const int m = m0 + mrow + r;
          const float val = acc[nt][r] + bias + (float)sh[(size_t)m * LDH + n];
          sh[(size_t)m * LDH + n] = (h16)val;
        }
      }
    }
    __syncthreads();
    ln_rows(sh, ln1_g + l * HID, ln1_b + l * HID, tid);
    __syncthreads();

    // FF1: f = relu(h @ W1^T + b1) -> sat
    {
      v8f acc[8];
#pragma unroll
      for (int i = 0; i < 8; ++i) acc[i] = vzero8();
      for (int k0 = 0; k0 < HID; k0 += 32) {
        v16h a = load_a16(sh, LDH, m0, k0, lane);
#pragma unroll
        for (int nt = 0; nt < 8; ++nt) {
          v16h b = load_b16(Wf1, HID, nt * 16, k0, lane);
          acc[nt] = wmma16(a, b, acc[nt]);
        }
      }
#pragma unroll
      for (int nt = 0; nt < 8; ++nt) {
        const int n = nt * 16 + ncol;
        const float bias = Bf1[n];
#pragma unroll
        for (int r = 0; r < 8; ++r)
          sat[(size_t)(m0 + mrow + r) * LDH + n] =
              (h16)fmaxf(acc[nt][r] + bias, 0.0f);
      }
    }
    __syncthreads();

    // FF2 + residual, then LN2
    {
      v8f acc[8];
#pragma unroll
      for (int i = 0; i < 8; ++i) acc[i] = vzero8();
      for (int k0 = 0; k0 < FFD; k0 += 32) {
        v16h a = load_a16(sat, LDH, m0, k0, lane);
#pragma unroll
        for (int nt = 0; nt < 8; ++nt) {
          v16h b = load_b16(Wf2, FFD, nt * 16, k0, lane);
          acc[nt] = wmma16(a, b, acc[nt]);
        }
      }
#pragma unroll
      for (int nt = 0; nt < 8; ++nt) {
        const int n = nt * 16 + ncol;
        const float bias = Bf2[n];
#pragma unroll
        for (int r = 0; r < 8; ++r) {
          const int m = m0 + mrow + r;
          const float val = acc[nt][r] + bias + (float)sh[(size_t)m * LDH + n];
          sh[(size_t)m * LDH + n] = (h16)val;
        }
      }
    }
    __syncthreads();
    ln_rows(sh, ln2_g + l * HID, ln2_b + l * HID, tid);
    __syncthreads();
  }

  // ---- Pooling over sequence -> pooled[s][:] ----
  if (tid < HID) {
    float acc = 0.f;
#pragma unroll 4
    for (int t = 0; t < KS; ++t) acc += (float)sh[(size_t)t * LDH + tid];
    pooled[(size_t)s * HID + tid] = acc * (1.0f / KS);
  }
}

// ---------------------------------------------------------------------------
// Final deterministic reduction + prediction MLP + sigmoid
// ---------------------------------------------------------------------------
__global__ __launch_bounds__(128) void finalize(
    const float* __restrict__ pooled, const float* __restrict__ w1,
    const float* __restrict__ b1, const float* __restrict__ w2,
    const float* __restrict__ b2, float* __restrict__ out) {
  __shared__ float fin[HID];
  __shared__ float hm[HID];
  const int n = threadIdx.x;
  float s = 0.f;
  for (int i = 0; i < DN; ++i) s += pooled[(size_t)i * HID + n];
  fin[n] = s * (1.0f / DN);
  __syncthreads();
  float a = b1[n];
  for (int k = 0; k < HID; ++k) a += fin[k] * w1[(size_t)n * HID + k];
  hm[n] = fmaxf(a, 0.0f);
  __syncthreads();
  if (n == 0) {
    float z = b2[0];
    for (int k = 0; k < HID; ++k) z += hm[k] * w2[k];
    out[0] = 1.0f / (1.0f + __expf(-z));
  }
}

// ---------------------------------------------------------------------------
extern "C" void kernel_launch(void* const* d_in, const int* in_sizes, int n_in,
                              void* d_out, int out_size, void* d_ws,
                              size_t ws_size, hipStream_t stream) {
  (void)in_sizes; (void)n_in; (void)out_size; (void)ws_size;
  const int* tgt = (const int*)d_in[0];
  const int* edge_index = (const int*)d_in[2];
  const float* edge_attrs = (const float*)d_in[3];
  const float* marker = (const float*)d_in[4];
  const float* proj_w = (const float*)d_in[5];
  const float* proj_b = (const float*)d_in[6];
  const float* attn_in_w = (const float*)d_in[7];
  const float* attn_in_b = (const float*)d_in[8];
  const float* attn_out_w = (const float*)d_in[9];
  const float* attn_out_b = (const float*)d_in[10];
  const float* ln1_g = (const float*)d_in[11];
  const float* ln1_b = (const float*)d_in[12];
  const float* ff1_w = (const float*)d_in[13];
  const float* ff1_b = (const float*)d_in[14];
  const float* ff2_w = (const float*)d_in[15];
  const float* ff2_b = (const float*)d_in[16];
  const float* ln2_g = (const float*)d_in[17];
  const float* ln2_b = (const float*)d_in[18];
  const float* pred_w1 = (const float*)d_in[19];
  const float* pred_b1 = (const float*)d_in[20];
  const float* pred_w2 = (const float*)d_in[21];
  const float* pred_b2 = (const float*)d_in[22];

  char* ws = (char*)d_ws;
  int* tgtEdge = (int*)ws;                              // DN ints
  float* pooled = (float*)(ws + 4096);                  // DN*HID f32
  h16* w16 = (h16*)(ws + 4096 + (size_t)DN * HID * 4);  // f16 weights
  h16* w_proj = w16;                                    // HID*ED
  h16* w_in = w_proj + HID * ED;                        // NL*3*HID*HID
  h16* w_out = w_in + NL * 3 * HID * HID;               // NL*HID*HID
  h16* w_ff1 = w_out + NL * HID * HID;                  // NL*FFD*HID
  h16* w_ff2 = w_ff1 + NL * FFD * HID;                  // NL*HID*FFD

  const int nproj = HID * ED;
  const int nin = NL * 3 * HID * HID;
  const int nsq = NL * HID * HID;
  f32_to_f16<<<(nproj + 255) / 256, 256, 0, stream>>>(proj_w, w_proj, nproj);
  f32_to_f16<<<(nin + 255) / 256, 256, 0, stream>>>(attn_in_w, w_in, nin);
  f32_to_f16<<<(nsq + 255) / 256, 256, 0, stream>>>(attn_out_w, w_out, nsq);
  f32_to_f16<<<(nsq + 255) / 256, 256, 0, stream>>>(ff1_w, w_ff1, nsq);
  f32_to_f16<<<(nsq + 255) / 256, 256, 0, stream>>>(ff2_w, w_ff2, nsq);

  find_tgt_edges<<<1, 1024, 0, stream>>>(edge_index, DN * KS, tgt, tgtEdge);

  (void)hipFuncSetAttribute((const void*)neighbor_transformer,
                            hipFuncAttributeMaxDynamicSharedMemorySize,
                            SMEM_BYTES);
  neighbor_transformer<<<DN, 256, SMEM_BYTES, stream>>>(
      edge_attrs, tgtEdge, w_proj, proj_b, marker, w_in, attn_in_b, w_out,
      attn_out_b, ln1_g, ln1_b, w_ff1, ff1_b, w_ff2, ff2_b, ln2_g, ln2_b,
      pooled);

  finalize<<<1, 128, 0, stream>>>(pooled, pred_w1, pred_b1, pred_w2, pred_b2,
                                  (float*)d_out);
}
